// Features_41180146434724
// MI455X (gfx1250) — matrix operand order repaired and checked by reading, not compiled, not run on hardware
//
#include <hip/hip_runtime.h>

// ---------------------------------------------------------------------------
// CDNA5 / gfx1250 feature-embedding kernels, bf16 WMMA (v_wmma_f32_16x16x32_bf16)
// ---------------------------------------------------------------------------

typedef __attribute__((ext_vector_type(16))) __bf16 v16bf;
typedef __attribute__((ext_vector_type(8)))  __bf16 v8bf;
typedef __attribute__((ext_vector_type(8)))  float  v8f;

#define LANE   (threadIdx.x & 31)
#define WAVE   (threadIdx.x >> 5)

__device__ __forceinline__ v8f wmma_bf16(v16bf a, v16bf b, v8f c) {
  return __builtin_amdgcn_wmma_f32_16x16x32_bf16(false, a, false, b, (short)0, c,
                                                 false, false);
}

__device__ __forceinline__ v8f bcast8(float x) {
  v8f c;
#pragma unroll
  for (int v = 0; v < 8; ++v) c[v] = x;
  return c;
}

__device__ __forceinline__ float gelu_f(float x) {
  return 0.5f * x * (1.0f + erff(x * 0.70710678118654752f));
}

// A-matrix fragment (16x32 bf16) from a row-major bf16 buffer (LDS or global).
// ISA layout: lane<16 holds row=lane, K in {0..7,16..23}; lane>=16 row=lane-16,
// K in {8..15, 24..31}.
__device__ __forceinline__ v16bf a_frag_bf16(const __bf16* base, int stride,
                                             int k0, int lane) {
  const __bf16* p = base + (size_t)(lane & 15) * stride + k0 + ((lane & 16) ? 8 : 0);
  v8bf lo = *(const v8bf*)p;
  v8bf hi = *(const v8bf*)(p + 16);
  v16bf a;
#pragma unroll
  for (int j = 0; j < 8; ++j) { a[j] = lo[j]; a[j + 8] = hi[j]; }
  return a;
}

// Same, sourcing f32 and converting on the fly.
__device__ __forceinline__ v16bf a_frag_f32(const float* base, int stride,
                                            int k0, int lane) {
  const float* p = base + (size_t)(lane & 15) * stride + k0 + ((lane & 16) ? 8 : 0);
  v16bf a;
#pragma unroll
  for (int j = 0; j < 8; ++j) {
    a[j]     = (__bf16)p[j];
    a[j + 8] = (__bf16)p[16 + j];
  }
  return a;
}

// B-matrix fragment (32x16 bf16) from transposed weights Wt[N][Kpad]:
// lane holds column N=lane&15, K in {8h..8h+7, 16+8h..23+8h} (h = lane/16).
__device__ __forceinline__ v16bf b_frag(const __bf16* Wt, int Kpad, int nb,
                                        int k0, int lane) {
  const __bf16* p = Wt + (size_t)(nb * 16 + (lane & 15)) * Kpad + k0 +
                    ((lane & 16) ? 8 : 0);
  v8bf lo = *(const v8bf*)p;
  v8bf hi = *(const v8bf*)(p + 16);
  v16bf b;
#pragma unroll
  for (int j = 0; j < 8; ++j) { b[j] = lo[j]; b[j + 8] = hi[j]; }
  return b;
}

// ---------------------------------------------------------------------------
// Prep: W (K x N, f32, row-major) -> Wt (N x Kpad, bf16), zero padding rows.
// ---------------------------------------------------------------------------
__global__ void transpose_bf16_kernel(const float* __restrict__ W,
                                      __bf16* __restrict__ Wt,
                                      int K, int N, int Kpad) {
  int idx = blockIdx.x * blockDim.x + threadIdx.x;
  if (idx >= N * Kpad) return;
  int n = idx / Kpad, k = idx - n * Kpad;
  Wt[idx] = (k < K) ? (__bf16)W[(size_t)k * N + n] : (__bf16)0.0f;
}

// ---------------------------------------------------------------------------
// Generic 2-layer MLP (K1 -> 128 -> 64) with exact GELU; one 16-row tile per
// wave, X staged to LDS once (bf16). Output written as bf16 into the [row][160]
// concat workspace at column colOff.
// ---------------------------------------------------------------------------
template <int K1>
__global__ void __launch_bounds__(128) mlp2_kernel(
    const float* __restrict__ X, const __bf16* __restrict__ W1t,
    const float* __restrict__ b1, const __bf16* __restrict__ W2t,
    const float* __restrict__ b2, __bf16* __restrict__ Cws, int colOff) {
  __shared__ __bf16 sX[4][16 * K1];
  __shared__ __bf16 sH[4][16 * 128];
  const int wave = WAVE, lane = LANE;
  const int r0 = (blockIdx.x * 4 + wave) * 16;
  const int col = lane & 15, rbase = (lane & 16) ? 8 : 0;
  __bf16* sx = sX[wave];
  __bf16* sh = sH[wave];

  // stage tile (16 contiguous rows of K1 floats) into LDS as bf16
  const float* Xt = X + (size_t)r0 * K1;
  for (int e = lane * 4; e < 16 * K1; e += 128) {
    float4 v = *(const float4*)(Xt + e);
    sx[e] = (__bf16)v.x; sx[e + 1] = (__bf16)v.y;
    sx[e + 2] = (__bf16)v.z; sx[e + 3] = (__bf16)v.w;
  }
  __syncthreads();

  v16bf afr[K1 / 32];
#pragma unroll
  for (int kk = 0; kk < K1 / 32; ++kk) afr[kk] = a_frag_bf16(sx, K1, kk * 32, lane);

  // layer 1: K1 -> 128, GELU, keep in LDS
  for (int nb = 0; nb < 8; ++nb) {
    v8f acc = bcast8(b1[nb * 16 + col]);
#pragma unroll
    for (int kk = 0; kk < K1 / 32; ++kk)
      acc = wmma_bf16(afr[kk], b_frag(W1t, K1, nb, kk * 32, lane), acc);
#pragma unroll
    for (int v = 0; v < 8; ++v)
      sh[(rbase + v) * 128 + nb * 16 + col] = (__bf16)gelu_f(acc[v]);
  }
  __syncthreads();

  v16bf hfr[4];
#pragma unroll
  for (int kk = 0; kk < 4; ++kk) hfr[kk] = a_frag_bf16(sh, 128, kk * 32, lane);

  // layer 2: 128 -> 64, write bf16 into concat workspace
  for (int nb = 0; nb < 4; ++nb) {
    v8f acc = bcast8(b2[nb * 16 + col]);
#pragma unroll
    for (int kk = 0; kk < 4; ++kk)
      acc = wmma_bf16(hfr[kk], b_frag(W2t, 128, nb, kk * 32, lane), acc);
#pragma unroll
    for (int v = 0; v < 8; ++v)
      Cws[(size_t)(r0 + rbase + v) * 160 + colOff + nb * 16 + col] = (__bf16)acc[v];
  }
}

// ---------------------------------------------------------------------------
// pae path: 16 -> 16 -> 16 (K padded to 32 with zeros). Writes cols 128..143
// and zeros cols 144..159 of the concat workspace.
// ---------------------------------------------------------------------------
__global__ void __launch_bounds__(128) pae_kernel(
    const float* __restrict__ X, const __bf16* __restrict__ W1t,
    const float* __restrict__ b1, const __bf16* __restrict__ W2t,
    const float* __restrict__ b2, __bf16* __restrict__ Cws) {
  __shared__ __bf16 sH[4][16 * 16];
  const int wave = WAVE, lane = LANE;
  const int r0 = (blockIdx.x * 4 + wave) * 16;
  const int col = lane & 15, rbase = (lane & 16) ? 8 : 0;
  __bf16* sh = sH[wave];

  v16bf a;
  {
    const float* p = X + (size_t)(r0 + (lane & 15)) * 16 + ((lane & 16) ? 8 : 0);
#pragma unroll
    for (int j = 0; j < 8; ++j) { a[j] = (__bf16)p[j]; a[j + 8] = (__bf16)0.0f; }
  }
  v8f acc = bcast8(b1[col]);
  acc = wmma_bf16(a, b_frag(W1t, 32, 0, 0, lane), acc);
#pragma unroll
  for (int v = 0; v < 8; ++v)
    sh[(rbase + v) * 16 + col] = (__bf16)gelu_f(acc[v]);
  __syncthreads();

  v16bf a2;
  {
    const __bf16* p = sh + (lane & 15) * 16 + ((lane & 16) ? 8 : 0);
#pragma unroll
    for (int j = 0; j < 8; ++j) { a2[j] = p[j]; a2[j + 8] = (__bf16)0.0f; }
  }
  v8f acc2 = bcast8(b2[col]);
  acc2 = wmma_bf16(a2, b_frag(W2t, 32, 0, 0, lane), acc2);
#pragma unroll
  for (int v = 0; v < 8; ++v) {
    size_t rr = (size_t)(r0 + rbase + v) * 160;
    Cws[rr + 128 + col] = (__bf16)acc2[v];
    Cws[rr + 144 + col] = (__bf16)0.0f;   // zero pad cols 144..159
  }
}

// ---------------------------------------------------------------------------
// Final pair projection: Y = (concat + relpos_emb[rel_idx]) @ po_W + po_b.
// relpos gather/clip fused into the A-fragment build over K=160 (pad = 0).
// Also writes pair_mask.
// ---------------------------------------------------------------------------
__device__ __forceinline__ v16bf a_frag_cat(const __bf16* crow, const float* prow,
                                            int k0, int lane) {
  const int khi = (lane & 16) ? 8 : 0;
  const int ka = k0 + khi, kb = ka + 16;
  v16bf a;
#pragma unroll
  for (int j = 0; j < 8; ++j) {
    int k1 = ka + j, k2 = kb + j;
    float v1 = (float)crow[k1] + ((k1 < 144) ? prow[k1] : 0.0f);
    float v2 = (float)crow[k2] + ((k2 < 144) ? prow[k2] : 0.0f);
    a[j] = (__bf16)v1;
    a[j + 8] = (__bf16)v2;
  }
  return a;
}

__global__ void __launch_bounds__(128) pair_out_kernel(
    const __bf16* __restrict__ Cws, const float* __restrict__ relpos,
    const int* __restrict__ chain, const int* __restrict__ seqtok,
    const __bf16* __restrict__ poWt, const float* __restrict__ pob,
    float* __restrict__ outPair, float* __restrict__ pMask) {
  const int wave = WAVE, lane = LANE;
  const int r0 = (blockIdx.x * 4 + wave) * 16;
  const int col = lane & 15, rbase = (lane & 16) ? 8 : 0;

  const int myr = r0 + (lane & 15);
  const int b = myr / 9216;
  const int rem = myr - b * 9216;
  const int i = rem / 96, j = rem - i * 96;
  const int posi = chain[b * 96 + i] * 100 + i;
  const int posj = chain[b * 96 + j] * 100 + j;
  int rel = posi - posj + 400;
  rel = rel < 0 ? 0 : (rel > 800 ? 800 : rel);

  const __bf16* crow = Cws + (size_t)myr * 160;
  const float* prow = relpos + (size_t)rel * 144;

  v16bf afr[5];
#pragma unroll
  for (int kk = 0; kk < 5; ++kk) afr[kk] = a_frag_cat(crow, prow, kk * 32, lane);

  for (int nb = 0; nb < 8; ++nb) {
    v8f acc = bcast8(pob[nb * 16 + col]);
#pragma unroll
    for (int kk = 0; kk < 5; ++kk)
      acc = wmma_bf16(afr[kk], b_frag(poWt, 160, nb, kk * 32, lane), acc);
#pragma unroll
    for (int v = 0; v < 8; ++v)
      outPair[(size_t)(r0 + rbase + v) * 128 + nb * 16 + col] = acc[v];
  }

  if (lane < 16) {
    float mi = (seqtok[b * 96 + i] != 0) ? 1.0f : 0.0f;
    float mj = (seqtok[b * 96 + j] != 0) ? 1.0f : 0.0f;
    pMask[myr] = mi * mj;
  }
}

// ---------------------------------------------------------------------------
// Single path: MLP(384->128->64) + concat[aa|sc|seq|chain|plddt] + sinusoid PE
// (computed on the fly), then 208(->224)x128 projection. Also single_mask.
// ---------------------------------------------------------------------------
__global__ void __launch_bounds__(128) single_kernel(
    const int* __restrict__ seqtok, const float* __restrict__ Xs,
    const int* __restrict__ chain, const float* __restrict__ plddts,
    const float* __restrict__ seqemb, const float* __restrict__ aa_emb,
    const __bf16* __restrict__ W1t, const float* __restrict__ b1,
    const __bf16* __restrict__ W2t, const float* __restrict__ b2,
    const float* __restrict__ chain_emb, const float* __restrict__ plddt_emb,
    const __bf16* __restrict__ soWt, const float* __restrict__ sob,
    float* __restrict__ outS, float* __restrict__ sMask) {
  __shared__ __bf16 sH[4][16 * 128];
  __shared__ __bf16 sC[4][16 * 224];
  const int wave = WAVE, lane = LANE;
  const int r0 = (blockIdx.x * 4 + wave) * 16;
  const int col = lane & 15, rbase = (lane & 16) ? 8 : 0;
  __bf16* sh = sH[wave];
  __bf16* sc = sC[wave];
  const float* Xrow0 = Xs + (size_t)r0 * 384;

  // layer 1: 384 -> 128, GELU (A re-fetched per nb; small problem, cache-hot)
  for (int nb = 0; nb < 8; ++nb) {
    v8f acc = bcast8(b1[nb * 16 + col]);
#pragma unroll
    for (int kk = 0; kk < 12; ++kk)
      acc = wmma_bf16(a_frag_f32(Xrow0, 384, kk * 32, lane),
                      b_frag(W1t, 384, nb, kk * 32, lane), acc);
#pragma unroll
    for (int v = 0; v < 8; ++v)
      sh[(rbase + v) * 128 + nb * 16 + col] = (__bf16)gelu_f(acc[v]);
  }
  __syncthreads();

  // layer 2: 128 -> 64 into concat cols 56..119
  for (int nb = 0; nb < 4; ++nb) {
    v8f acc = bcast8(b2[nb * 16 + col]);
#pragma unroll
    for (int kk = 0; kk < 4; ++kk)
      acc = wmma_bf16(a_frag_bf16(sh, 128, kk * 32, lane),
                      b_frag(W2t, 128, nb, kk * 32, lane), acc);
#pragma unroll
    for (int v = 0; v < 8; ++v)
      sc[(rbase + v) * 224 + 56 + nb * 16 + col] = (__bf16)acc[v];
  }
  __syncthreads();

  // fill remaining concat segments + add sinusoidal positional encoding
  for (int e = lane; e < 16 * 224; e += 32) {
    int row = e / 224, k = e - row * 224;
    int r = r0 + row;
    int bb = r / 96, ii = r - bb * 96;
    float v;
    if (k < 56) {
      v = aa_emb[seqtok[r] * 56 + k];
    } else if (k < 120) {
      v = (float)sc[e];                       // single_c already placed
    } else if (k < 184) {
      v = seqemb[(size_t)r * 64 + (k - 120)];
    } else if (k < 192) {
      v = chain_emb[chain[r] * 8 + (k - 184)];
    } else if (k < 208) {
      float pl = plddts[r];
      int bin = 0;
#pragma unroll
      for (int t = 0; t < 20; ++t) bin += (pl > (100.0f * t) / 19.0f) ? 1 : 0;
      if (bin > 19) bin = 19;
      v = plddt_emb[bin * 16 + (k - 192)];
    } else {
      v = 0.0f;                               // pad 208..223
    }
    if (k < 208) {
      int pos = chain[r] * 100 + ii;
      int i2 = k >> 1;
      float div = expf((float)(2 * i2) * (-9.210340371976184f / 208.0f));
      float ang = (float)pos * div;
      v += (k & 1) ? cosf(ang) : sinf(ang);
    }
    sc[e] = (__bf16)v;
  }
  __syncthreads();

  // output projection: 224 -> 128
  for (int nb = 0; nb < 8; ++nb) {
    v8f acc = bcast8(sob[nb * 16 + col]);
#pragma unroll
    for (int kk = 0; kk < 7; ++kk)
      acc = wmma_bf16(a_frag_bf16(sc, 224, kk * 32, lane),
                      b_frag(soWt, 224, nb, kk * 32, lane), acc);
#pragma unroll
    for (int v = 0; v < 8; ++v)
      outS[(size_t)(r0 + rbase + v) * 128 + nb * 16 + col] = acc[v];
  }

  if (lane < 16) {
    int r = r0 + lane;
    sMask[r] = (seqtok[r] != 0) ? 1.0f : 0.0f;
  }
}

// ---------------------------------------------------------------------------
// Point path: concat of 5 token embeddings (192) @ pt_W -> (32,128)
// ---------------------------------------------------------------------------
__global__ void __launch_bounds__(64) point_kernel(
    const int* __restrict__ va, const int* __restrict__ ja,
    const int* __restrict__ vb, const int* __restrict__ jb,
    const int* __restrict__ hla, const float* __restrict__ vaE,
    const float* __restrict__ jaE, const float* __restrict__ vbE,
    const float* __restrict__ jbE, const float* __restrict__ hlaE,
    const __bf16* __restrict__ ptWt, const float* __restrict__ ptb,
    float* __restrict__ outP) {
  __shared__ __bf16 sC[2][16 * 192];
  const int wave = WAVE, lane = LANE;
  const int r0 = wave * 16;
  __bf16* sc = sC[wave];

  for (int e = lane; e < 16 * 192; e += 32) {
    int row = e / 192, k = e - row * 192;
    int bb = r0 + row;
    float v;
    if (k < 32)       v = vaE[va[bb] * 32 + k];
    else if (k < 64)  v = jaE[ja[bb] * 32 + (k - 32)];
    else if (k < 96)  v = vbE[vb[bb] * 32 + (k - 64)];
    else if (k < 128) v = jbE[jb[bb] * 32 + (k - 96)];
    else              v = hlaE[hla[bb] * 64 + (k - 128)];
    sc[e] = (__bf16)v;
  }
  __syncthreads();

  const int col = lane & 15, rbase = (lane & 16) ? 8 : 0;
  for (int nb = 0; nb < 8; ++nb) {
    v8f acc = bcast8(ptb[nb * 16 + col]);
#pragma unroll
    for (int kk = 0; kk < 6; ++kk)
      acc = wmma_bf16(a_frag_bf16(sc, 192, kk * 32, lane),
                      b_frag(ptWt, 192, nb, kk * 32, lane), acc);
#pragma unroll
    for (int v = 0; v < 8; ++v)
      outP[(size_t)(r0 + rbase + v) * 128 + nb * 16 + col] = acc[v];
  }
}

// ---------------------------------------------------------------------------
// Host launcher
// ---------------------------------------------------------------------------
extern "C" void kernel_launch(void* const* d_in, const int* in_sizes, int n_in,
                              void* d_out, int out_size, void* d_ws,
                              size_t ws_size, hipStream_t stream) {
  (void)in_sizes; (void)n_in; (void)out_size; (void)ws_size;

  // ---- inputs (setup_inputs dict order; params flattened in insertion order)
  const int*   seqtok     = (const int*)d_in[0];
  const float* emb_single = (const float*)d_in[1];
  const float* emb_pair   = (const float*)d_in[2];
  const int*   chain      = (const int*)d_in[3];
  const float* dist       = (const float*)d_in[4];
  const float* pae        = (const float*)d_in[5];
  const float* plddts     = (const float*)d_in[6];
  const int*   va         = (const int*)d_in[7];
  const int*   ja         = (const int*)d_in[8];
  const int*   vb         = (const int*)d_in[9];
  const int*   jb         = (const int*)d_in[10];
  const int*   hla        = (const int*)d_in[11];
  const float* seq_embed  = (const float*)d_in[12];
  const float* aa_emb     = (const float*)d_in[13];
  const float* sc_W1 = (const float*)d_in[14]; const float* sc_b1 = (const float*)d_in[15];
  const float* sc_W2 = (const float*)d_in[16]; const float* sc_b2 = (const float*)d_in[17];
  const float* pc_W1 = (const float*)d_in[18]; const float* pc_b1 = (const float*)d_in[19];
  const float* pc_W2 = (const float*)d_in[20]; const float* pc_b2 = (const float*)d_in[21];
  const float* dc_W1 = (const float*)d_in[22]; const float* dc_b1 = (const float*)d_in[23];
  const float* dc_W2 = (const float*)d_in[24]; const float* dc_b2 = (const float*)d_in[25];
  const float* pa_W1 = (const float*)d_in[26]; const float* pa_b1 = (const float*)d_in[27];
  const float* pa_W2 = (const float*)d_in[28]; const float* pa_b2 = (const float*)d_in[29];
  const float* chain_emb  = (const float*)d_in[30];
  const float* plddt_emb  = (const float*)d_in[31];
  const float* va_emb = (const float*)d_in[32]; const float* ja_emb = (const float*)d_in[33];
  const float* vb_emb = (const float*)d_in[34]; const float* jb_emb = (const float*)d_in[35];
  const float* hla_emb    = (const float*)d_in[36];
  const float* relpos_emb = (const float*)d_in[37];
  const float* so_W = (const float*)d_in[38]; const float* so_b = (const float*)d_in[39];
  const float* po_W = (const float*)d_in[40]; const float* po_b = (const float*)d_in[41];
  const float* pt_W = (const float*)d_in[42]; const float* pt_b = (const float*)d_in[43];

  float* out = (float*)d_out;
  // output layout (floats): single | pair | point | single_mask | pair_mask
  const size_t OFF_PAIR  = 393216;      // 32*96*128
  const size_t OFF_POINT = OFF_PAIR + (size_t)294912 * 128;
  const size_t OFF_SMASK = OFF_POINT + 4096;
  const size_t OFF_PMASK = OFF_SMASK + 3072;

  // ---- workspace layout
  __bf16* Cws   = (__bf16*)d_ws;                       // 294912 x 160 bf16 concat
  __bf16* wbase = Cws + (size_t)294912 * 160;
  __bf16* pcW1t = wbase;                               // [128][128]
  __bf16* pcW2t = pcW1t + 128 * 128;                   // [64][128]
  __bf16* dcW1t = pcW2t + 64 * 128;                    // [128][256]
  __bf16* dcW2t = dcW1t + 128 * 256;                   // [64][128]
  __bf16* paW1t = dcW2t + 64 * 128;                    // [16][32]
  __bf16* paW2t = paW1t + 16 * 32;                     // [16][32]
  __bf16* poWt  = paW2t + 16 * 32;                     // [128][160]
  __bf16* scW1t = poWt + 128 * 160;                    // [128][384]
  __bf16* scW2t = scW1t + 128 * 384;                   // [64][128]
  __bf16* soWt  = scW2t + 64 * 128;                    // [128][224]
  __bf16* ptWt  = soWt + 128 * 224;                    // [128][192]

  // ---- prep: weights -> transposed bf16 (L2-resident thereafter)
  auto T = [&](const float* W, __bf16* Wt, int K, int N, int Kpad) {
    int elems = N * Kpad;
    transpose_bf16_kernel<<<(elems + 255) / 256, 256, 0, stream>>>(W, Wt, K, N, Kpad);
  };
  T(pc_W1, pcW1t, 128, 128, 128);
  T(pc_W2, pcW2t, 128, 64, 128);
  T(dc_W1, dcW1t, 256, 128, 256);
  T(dc_W2, dcW2t, 128, 64, 128);
  T(pa_W1, paW1t, 16, 16, 32);
  T(pa_W2, paW2t, 16, 16, 32);
  T(po_W, poWt, 144, 128, 160);
  T(sc_W1, scW1t, 384, 128, 384);
  T(sc_W2, scW2t, 128, 64, 128);
  T(so_W, soWt, 208, 128, 224);
  T(pt_W, ptWt, 192, 128, 192);

  // ---- pair-path MLPs into concat workspace (18432 tiles of 16 rows)
  mlp2_kernel<128><<<4608, 128, 0, stream>>>(emb_pair, pcW1t, pc_b1, pcW2t, pc_b2,
                                             Cws, 0);
  mlp2_kernel<256><<<4608, 128, 0, stream>>>(dist, dcW1t, dc_b1, dcW2t, dc_b2,
                                             Cws, 64);
  pae_kernel<<<4608, 128, 0, stream>>>(pae, paW1t, pa_b1, paW2t, pa_b2, Cws);

  // ---- final pair projection (+relpos, +mask)
  pair_out_kernel<<<4608, 128, 0, stream>>>(Cws, relpos_emb, chain, seqtok, poWt,
                                            po_b, out + OFF_PAIR, out + OFF_PMASK);

  // ---- single path (192 tiles) and point path
  single_kernel<<<48, 128, 0, stream>>>(seqtok, emb_single, chain, plddts,
                                        seq_embed, aa_emb, scW1t, sc_b1, scW2t,
                                        sc_b2, chain_emb, plddt_emb, soWt, so_b,
                                        out, out + OFF_SMASK);
  point_kernel<<<1, 64, 0, stream>>>(va, ja, vb, jb, hla, va_emb, ja_emb, vb_emb,
                                     jb_emb, hla_emb, ptWt, pt_b, out + OFF_POINT);
}
